// RecurrentRGCN_48215302865400
// MI455X (gfx1250) — matrix operand order repaired and checked by reading, not compiled, not run on hardware
//
#include <hip/hip_runtime.h>
#include <math.h>

// ---------------- problem constants (match reference) ----------------
#define N_ENTS   50000
#define N_RELS2  460
#define HD       128
#define T_STEPS  8
#define E_EDGES  200000
#define M_RTE    400000
#define RRELU_SLOPE 0.22916666666666666f
#define EPSV     1e-12f
#define RM_CHUNKS 64

// ---------------- WMMA types ----------------
typedef __attribute__((ext_vector_type(16))) __bf16 v16bf;
typedef __attribute__((ext_vector_type(8)))  float  v8f;

struct alignas(16) U128 { unsigned x, y, z, w; };
struct BF16x16 { U128 lo; U128 hi; };   // 32 bytes == v16bf

__device__ inline unsigned short f32_to_bf16_bits(float f) {
  unsigned u = __builtin_bit_cast(unsigned, f);
  u = u + 0x7FFFu + ((u >> 16) & 1u);           // round-to-nearest-even
  return (unsigned short)(u >> 16);
}

// ---------------- elementwise utility kernels ----------------
__global__ void k_zero(float* p, int n) {
  int i = blockIdx.x * blockDim.x + threadIdx.x;
  if (i < n) p[i] = 0.0f;
}

__global__ void k_copy(const float* s, float* d, int n) {
  int i = blockIdx.x * blockDim.x + threadIdx.x;
  if (i < n) d[i] = s[i];
}

// one block (128 thr) per row: out = row / max(||row||, eps)
__global__ void k_l2norm_rows(const float* in, float* out, int nrows) {
  __shared__ float red[HD];
  int r = blockIdx.x;
  if (r >= nrows) return;
  int j = threadIdx.x;
  float v = in[r * HD + j];
  red[j] = v * v;
  __syncthreads();
  for (int s = 64; s > 0; s >>= 1) { if (j < s) red[j] += red[j + s]; __syncthreads(); }
  float d = fmaxf(sqrtf(red[0]), EPSV);
  out[r * HD + j] = v / d;
}

// ---------------- relation mean: two-level segment sum ----------------
// grid = (RM_CHUNKS, 4 col-groups of 32), block = 256
__global__ void k_relmean(const int* __restrict__ rte, const int* __restrict__ rseg,
                          const float* __restrict__ h, float* sums, float* cnts) {
  __shared__ float lsum[N_RELS2 * 32];   // 58,880 B
  __shared__ float lcnt[N_RELS2];
  int cg = blockIdx.y;
  int chunk = blockIdx.x;
  int t = threadIdx.x, warp = t >> 5, lane = t & 31;
  for (int i = t; i < N_RELS2 * 32; i += 256) lsum[i] = 0.0f;
  if (cg == 0) for (int i = t; i < N_RELS2; i += 256) lcnt[i] = 0.0f;
  __syncthreads();
  const int per = M_RTE / RM_CHUNKS;     // 6250, exact
  int m0 = chunk * per, m1 = m0 + per;
  for (int m = m0 + warp; m < m1; m += 8) {
    int rel = rseg[m];
    int ent = rte[m];
    float v = h[ent * HD + cg * 32 + lane];
    atomicAdd(&lsum[rel * 32 + lane], v);
    if (cg == 0 && lane == 0) atomicAdd(&lcnt[rel], 1.0f);
  }
  __syncthreads();
  for (int i = t; i < N_RELS2 * 32; i += 256) {
    float v = lsum[i];
    if (v != 0.0f) atomicAdd(&sums[(i >> 5) * HD + cg * 32 + (i & 31)], v);
  }
  if (cg == 0)
    for (int r = t; r < N_RELS2; r += 256) {
      float c = lcnt[r];
      if (c != 0.0f) atomicAdd(&cnts[r], c);
    }
}

// in-place: sums -> mean (where cnt>0)
__global__ void k_xmean(float* sums, const float* cnts) {
  int i = blockIdx.x * blockDim.x + threadIdx.x;
  if (i < N_RELS2 * HD) {
    float c = cnts[i >> 7];
    sums[i] = (c > 0.0f) ? sums[i] / fmaxf(c, 1.0f) : 0.0f;
  }
}

// ---------------- GRU cell + l2norm (460 rows, tiny) ----------------
// Wiht: [256][384] = W_ih^T ; Whht: [128][384] = W_hh^T (pre-transposed)
__global__ void k_gru(const float* __restrict__ emb_rel, const float* __restrict__ xmean,
                      float* h0, const float* __restrict__ Wiht, const float* __restrict__ Whht,
                      const float* __restrict__ bih, const float* __restrict__ bhh) {
  __shared__ float xb[2 * HD];
  __shared__ float hb[HD];
  __shared__ float red[HD];
  int r = blockIdx.x, j = threadIdx.x;
  xb[j]      = emb_rel[r * HD + j];
  xb[HD + j] = xmean[r * HD + j];
  float hj = h0[r * HD + j];
  hb[j] = hj;
  __syncthreads();
  float ir = bih[j], iz = bih[HD + j], in_ = bih[2 * HD + j];
  for (int k = 0; k < 2 * HD; k++) {
    float xv = xb[k];
    const float* w = Wiht + k * 384;
    ir = fmaf(xv, w[j], ir); iz = fmaf(xv, w[HD + j], iz); in_ = fmaf(xv, w[2 * HD + j], in_);
  }
  float hr = bhh[j], hz = bhh[HD + j], hn = bhh[2 * HD + j];
  for (int k = 0; k < HD; k++) {
    float hv = hb[k];
    const float* w = Whht + k * 384;
    hr = fmaf(hv, w[j], hr); hz = fmaf(hv, w[HD + j], hz); hn = fmaf(hv, w[2 * HD + j], hn);
  }
  float rg = 1.0f / (1.0f + __expf(-(ir + hr)));
  float zg = 1.0f / (1.0f + __expf(-(iz + hz)));
  float ng = tanhf(in_ + rg * hn);
  float hv = (1.0f - zg) * ng + zg * hj;
  red[j] = hv * hv;
  __syncthreads();
  for (int s = 64; s > 0; s >>= 1) { if (j < s) red[j] += red[j + s]; __syncthreads(); }
  float d = fmaxf(sqrtf(red[0]), EPSV);
  h0[r * HD + j] = hv / d;
}

// ---------------- degree / norm ----------------
__global__ void k_deg(const int* __restrict__ dst, float* deg) {
  int e = blockIdx.x * blockDim.x + threadIdx.x;
  if (e < E_EDGES) atomicAdd(&deg[dst[e]], 1.0f);
}
__global__ void k_invdeg(float* deg) {
  int i = blockIdx.x * blockDim.x + threadIdx.x;
  if (i < N_ENTS) { float d = deg[i]; deg[i] = (d > 0.0f) ? 1.0f / d : 0.0f; }
}

// ---------------- weight prep ----------------
// Pack W (128x128 row-major, D = A@W) into per-lane WMMA B-fragments (bf16).
// frag[((nt*4+ks)*32+lane)*16 + e] = W[k][n] with
//   n = nt*16 + (lane&15),  k = ks*32 + (lane>=16 ? 16 : 0) + e    (contiguous-K order, same as A frags)
__global__ void k_prep_frags(const float* __restrict__ rgcnW, const float* __restrict__ tgw,
                             unsigned short* fragRG, unsigned short* fragTG) {
  int idx = blockIdx.x * blockDim.x + threadIdx.x;
  if (idx >= 3 * 16384) return;
  int mat = idx >> 14, r = idx & 16383;
  int nt = r >> 11, ks = (r >> 9) & 3, lane = (r >> 4) & 31, e = r & 15;
  int n = nt * 16 + (lane & 15);
  int k = ks * 32 + ((lane >> 4) * 16) + e;
  const float* W = (mat == 2) ? tgw : (rgcnW + mat * 16384);
  unsigned short* d = (mat == 2) ? fragTG : (fragRG + mat * 16384);
  d[r] = f32_to_bf16_bits(W[k * HD + n]);
}

__global__ void k_prep_wt(const float* __restrict__ Wih, const float* __restrict__ Whh,
                          float* Wiht, float* Whht) {
  int idx = blockIdx.x * blockDim.x + threadIdx.x;
  if (idx < 256 * 384) {
    int k = idx / 384, j = idx % 384;
    Wiht[idx] = Wih[j * 256 + k];
  } else if (idx < 256 * 384 + 128 * 384) {
    int r = idx - 256 * 384;
    int k = r / 384, j = r % 384;
    Whht[r] = Whh[j * 128 + k];
  }
}

// ---------------- shared WMMA tile core ----------------
// A staged in LDS as bf16[16][136] (272B pitch -> 4-bank skew, no conflicts).
// A-frag (contiguous K, matches B frags): lane<16: elems0-15 = K k0..k0+15 (row=lane)
//                             lane>=16: shifted by +16 (row=lane-16)
__device__ inline void wmma_tile_128(const unsigned short* A, const unsigned short* frag,
                                     int lane, v8f acc[8]) {
  int arow = lane & 15;
  int khalf = (lane >> 4) << 4;
#pragma unroll
  for (int ks = 0; ks < 4; ks++) {
    int kb = ks * 32 + khalf;
    BF16x16 t;
    t.lo = *(const U128*)&A[arow * 136 + kb];
    t.hi = *(const U128*)&A[arow * 136 + kb + 8];
    v16bf a = __builtin_bit_cast(v16bf, t);
#pragma unroll
    for (int nt = 0; nt < 8; nt++) {
      v16bf b = *(const v16bf*)(frag + ((nt * 4 + ks) * 32 + lane) * 16);
      acc[nt] = __builtin_amdgcn_wmma_f32_16x16x32_bf16(
          false, a, false, b, (short)0, acc[nt], false, false);
    }
  }
}

// ---------------- RGCN layer: msg = (nh[src]+h0[etype]) @ W, scatter-add by dst ----------------
__global__ void __launch_bounds__(256) k_rgcn_gemm(
    const int* __restrict__ src, const int* __restrict__ ety, const int* __restrict__ dst,
    const float* __restrict__ nh, const float* __restrict__ h0,
    const unsigned short* __restrict__ frag, float* agg, int ntiles) {
  __shared__ __align__(16) unsigned short at[8 * 16 * 136];
  int warp = threadIdx.x >> 5, lane = threadIdx.x & 31;
  int tile = blockIdx.x * 8 + warp;
  if (tile >= ntiles) return;                 // wave-uniform, EXEC stays full
  unsigned short* A = at + warp * (16 * 136);
  int e0 = tile * 16;
  for (int i = lane; i < 2048; i += 32) {
    int row = i >> 7, col = i & 127;
    int s = src[e0 + row];
    int rr = ety[e0 + row];
    float v = nh[s * HD + col] + h0[rr * HD + col];
    A[row * 136 + col] = f32_to_bf16_bits(v);
  }
  v8f acc[8] = {};
  wmma_tile_128(A, frag, lane, acc);
  int rlo = (lane >> 4) * 8;
  int nc = lane & 15;
  int dcache[8];
#pragma unroll
  for (int i = 0; i < 8; i++) dcache[i] = dst[e0 + rlo + i];
#pragma unroll
  for (int nt = 0; nt < 8; nt++) {
#pragma unroll
    for (int i = 0; i < 8; i++)
      atomicAdd(&agg[dcache[i] * HD + nt * 16 + nc], acc[nt][i]);
  }
}

// per-entity: v = rrelu(agg*norm); optional row l2norm
__global__ void k_rgcn_finalize(const float* __restrict__ agg, const float* __restrict__ norm,
                                float* out, int do_l2) {
  __shared__ float red[HD];
  int r = blockIdx.x, j = threadIdx.x;
  float v = agg[r * HD + j] * norm[r];
  v = (v >= 0.0f) ? v : RRELU_SLOPE * v;
  if (do_l2) {
    red[j] = v * v;
    __syncthreads();
    for (int s = 64; s > 0; s >>= 1) { if (j < s) red[j] += red[j + s]; __syncthreads(); }
    float d = fmaxf(sqrtf(red[0]), EPSV);
    v = v / d;
  }
  out[r * HD + j] = v;
}

// ---------------- time gate: tw = sigmoid(h@TGW + b); out = tw*cur + (1-tw)*h ----------------
__global__ void __launch_bounds__(256) k_timegate(
    const float* __restrict__ hprev, const float* __restrict__ cur,
    const unsigned short* __restrict__ frag, const float* __restrict__ bias,
    float* hout, int ntiles) {
  __shared__ __align__(16) unsigned short at[8 * 16 * 136];
  int warp = threadIdx.x >> 5, lane = threadIdx.x & 31;
  int tile = blockIdx.x * 8 + warp;
  if (tile >= ntiles) return;
  unsigned short* A = at + warp * (16 * 136);
  int r0 = tile * 16;
  for (int i = lane; i < 2048; i += 32) {
    int row = i >> 7, col = i & 127;
    A[row * 136 + col] = f32_to_bf16_bits(hprev[(r0 + row) * HD + col]);
  }
  v8f acc[8] = {};
  wmma_tile_128(A, frag, lane, acc);
  int rlo = (lane >> 4) * 8;
  int nc = lane & 15;
#pragma unroll
  for (int nt = 0; nt < 8; nt++) {
    int col = nt * 16 + nc;
    float b = bias[col];
#pragma unroll
    for (int i = 0; i < 8; i++) {
      int row = r0 + rlo + i;
      float g = acc[nt][i] + b;
      float tw = 1.0f / (1.0f + __expf(-g));
      float hp = hprev[row * HD + col];
      float cv = cur[row * HD + col];
      hout[row * HD + col] = tw * cv + (1.0f - tw) * hp;
    }
  }
}

// ---------------- host driver ----------------
extern "C" void kernel_launch(void* const* d_in, const int* in_sizes, int n_in,
                              void* d_out, int out_size, void* d_ws, size_t ws_size,
                              hipStream_t stream) {
  (void)in_sizes; (void)n_in; (void)out_size; (void)ws_size;
  const int*   src    = (const int*)d_in[0];
  const int*   dst    = (const int*)d_in[1];
  const int*   ety    = (const int*)d_in[2];
  const int*   rte    = (const int*)d_in[3];
  const int*   rseg   = (const int*)d_in[4];
  const float* demb   = (const float*)d_in[5];
  const float* embrel = (const float*)d_in[6];
  const float* Wih    = (const float*)d_in[7];
  const float* Whh    = (const float*)d_in[8];
  const float* bih    = (const float*)d_in[9];
  const float* bhh    = (const float*)d_in[10];
  const float* rgcnW  = (const float*)d_in[11];
  const float* tgw    = (const float*)d_in[12];
  const float* tgb    = (const float*)d_in[13];
  float* out = (float*)d_out;

  char* ws = (char*)d_ws;
  size_t off = 0;
  auto alloc = [&](size_t bytes) -> char* {
    char* p = ws + off;
    off += (bytes + 255) & ~(size_t)255;
    return p;
  };
  float* h_init = (float*)alloc((size_t)N_ENTS * HD * 4);
  float* agg    = (float*)alloc((size_t)N_ENTS * HD * 4);
  float* nh1    = (float*)alloc((size_t)N_ENTS * HD * 4);
  float* normb  = (float*)alloc((size_t)N_ENTS * 4);
  float* sums   = (float*)alloc((size_t)N_RELS2 * HD * 4);
  float* cnts   = (float*)alloc((size_t)N_RELS2 * 4);
  float* h0     = (float*)alloc((size_t)N_RELS2 * HD * 4);
  unsigned short* fragRG = (unsigned short*)alloc(2 * 16384 * 2);
  unsigned short* fragTG = (unsigned short*)alloc(16384 * 2);
  float* Wiht = (float*)alloc(256 * 384 * 4);
  float* Whht = (float*)alloc(128 * 384 * 4);

  // one-time (per call) prep: weight fragments, transposes, initial states
  k_prep_frags<<<(3 * 16384 + 255) / 256, 256, 0, stream>>>(rgcnW, tgw, fragRG, fragTG);
  k_prep_wt<<<((256 * 384 + 128 * 384) + 255) / 256, 256, 0, stream>>>(Wih, Whh, Wiht, Whht);
  k_l2norm_rows<<<N_ENTS, HD, 0, stream>>>(demb, h_init, N_ENTS);
  k_copy<<<(N_RELS2 * HD + 255) / 256, 256, 0, stream>>>(embrel, h0, N_RELS2 * HD);

  const int ETILES = E_EDGES / 16;   // 12500
  const int NTILES = N_ENTS / 16;    // 3125

  for (int t = 0; t < T_STEPS; t++) {
    const float* h = (t == 0) ? h_init : (out + (size_t)(t - 1) * N_ENTS * HD);
    float* hout = out + (size_t)t * N_ENTS * HD;
    const int* src_t = src + t * E_EDGES;
    const int* dst_t = dst + t * E_EDGES;
    const int* ety_t = ety + t * E_EDGES;
    const int* rte_t = rte + t * M_RTE;
    const int* rseg_t = rseg + t * M_RTE;

    // relation evolution
    k_zero<<<(N_RELS2 * HD + 255) / 256, 256, 0, stream>>>(sums, N_RELS2 * HD);
    k_zero<<<(N_RELS2 + 255) / 256, 256, 0, stream>>>(cnts, N_RELS2);
    dim3 rg(RM_CHUNKS, 4);
    k_relmean<<<rg, 256, 0, stream>>>(rte_t, rseg_t, h, sums, cnts);
    k_xmean<<<(N_RELS2 * HD + 255) / 256, 256, 0, stream>>>(sums, cnts);
    k_gru<<<N_RELS2, HD, 0, stream>>>(embrel, sums, h0, Wiht, Whht, bih, bhh);

    // in-degree norm
    k_zero<<<(N_ENTS + 255) / 256, 256, 0, stream>>>(normb, N_ENTS);
    k_deg<<<(E_EDGES + 255) / 256, 256, 0, stream>>>(dst_t, normb);
    k_invdeg<<<(N_ENTS + 255) / 256, 256, 0, stream>>>(normb);

    // RGCN layer 0
    k_zero<<<(N_ENTS * HD + 255) / 256, 256, 0, stream>>>(agg, N_ENTS * HD);
    k_rgcn_gemm<<<(ETILES + 7) / 8, 256, 0, stream>>>(src_t, ety_t, dst_t, h, h0, fragRG, agg, ETILES);
    k_rgcn_finalize<<<N_ENTS, HD, 0, stream>>>(agg, normb, nh1, 0);
    // RGCN layer 1
    k_zero<<<(N_ENTS * HD + 255) / 256, 256, 0, stream>>>(agg, N_ENTS * HD);
    k_rgcn_gemm<<<(ETILES + 7) / 8, 256, 0, stream>>>(src_t, ety_t, dst_t, nh1, h0, fragRG + 16384, agg, ETILES);
    k_rgcn_finalize<<<N_ENTS, HD, 0, stream>>>(agg, normb, nh1, 1);   // nh1 := current_h

    // time gate fusion -> history slice (also next step's h)
    k_timegate<<<(NTILES + 7) / 8, 256, 0, stream>>>(h, nh1, fragTG, tgb, hout, NTILES);
  }

  // final h0 appended after history
  k_copy<<<(N_RELS2 * HD + 255) / 256, 256, 0, stream>>>(
      h0, out + (size_t)T_STEPS * N_ENTS * HD, N_RELS2 * HD);
}
